// SemiCRF_65661460021838
// MI455X (gfx1250) — matrix operand order repaired and checked by reading, not compiled, not run on hardware
//
#include <hip/hip_runtime.h>
#include <hip/hip_bf16.h>
#include <math.h>

// ---------------------------------------------------------------------------
// SemiCRF on MI455X (gfx1250, wave32, WMMA).
// Heavy math folded into ONE f16 WMMA GEMM: (B*S=8192) x (D=512) x (2304),
// ~9.7 G MACs, followed by a cheap tap-gather and a latency-bound LDS scan.
// GEMM v2: B tile stored fragment-contiguous in LDS (2x ds_load_b128 per
// fragment instead of 16x ds_load_u16 + packing), double-buffered LDS with
// global loads issued ahead of the WMMA block.
// ---------------------------------------------------------------------------

typedef __attribute__((ext_vector_type(16))) _Float16 v16h;
typedef __attribute__((ext_vector_type(8)))  float    v8f;

#define NEGV  (-10000.0f)
#define Bdim  32
#define Sdim  256
#define Ddim  512
#define Kseg  8
#define Ldim  64
#define NPAIR 36            // lower-triangular (k,j) tap pairs, j<=k
#define Ncols (NPAIR*Ldim)  // 2304 GEMM columns
#define Mrows (Bdim*Sdim)   // 8192 GEMM rows

// ============================================================================
// Kernel 1: W2[p=(k,j), l, d] = sum_dout cls_w[l,dout] * conv_w[k,dout,d,j]
// ============================================================================
__global__ __launch_bounds__(256) void w2_kernel(
    const float* __restrict__ conv_w,   // (K, D, D, K) = [k][dout][din][j]
    const float* __restrict__ cls_w,    // (L, D)
    _Float16* __restrict__ W2h)         // (Ncols, D) row-major
{
    int p    = blockIdx.x;       // 0..35
    int dblk = blockIdx.y;       // 0..7 : din block of 64
    int k = 0, rem = p;
    while (rem > k) { rem -= (k + 1); ++k; }
    int j = rem;                 // pair p -> (k, j)

    __shared__ float cls_sh[64][65];    // (l, dout)
    __shared__ float conv_sh[64][65];   // (dout, din)

    int t    = threadIdx.x;
    int lq   = t >> 2;           // 0..63  (loader row)
    int qi   = (t & 3) * 16;     // loader col group
    int l    = t & 63;           // compute: label
    int dsub = (t >> 6) * 16;    // compute: 16 consecutive din
    int din0 = dblk * 64;

    float acc[16];
#pragma unroll
    for (int i = 0; i < 16; ++i) acc[i] = 0.0f;

    for (int dout0 = 0; dout0 < Ddim; dout0 += 64) {
        __syncthreads();
#pragma unroll
        for (int i = 0; i < 16; ++i)
            cls_sh[lq][qi + i] = cls_w[lq * Ddim + dout0 + qi + i];
#pragma unroll
        for (int i = 0; i < 16; ++i)
            conv_sh[lq][qi + i] =
                conv_w[(((size_t)k * Ddim + dout0 + lq) * Ddim + din0 + qi + i) * Kseg + j];
        __syncthreads();
        for (int dd = 0; dd < 64; ++dd) {
            float c = cls_sh[l][dd];
#pragma unroll
            for (int i = 0; i < 16; ++i)
                acc[i] = fmaf(c, conv_sh[dd][dsub + i], acc[i]);
        }
    }
#pragma unroll
    for (int i = 0; i < 16; ++i)
        W2h[(size_t)(p * Ldim + l) * Ddim + din0 + dsub + i] = (_Float16)acc[i];
}

// ============================================================================
// Kernel 2: bias2[k,l] = cls_b[l] + cls_w[l,:]·conv_b[k,:]
// ============================================================================
__global__ void bias2_kernel(const float* __restrict__ cls_w,
                             const float* __restrict__ cls_b,
                             const float* __restrict__ conv_b,
                             float* __restrict__ bias2)
{
    int idx = blockIdx.x * blockDim.x + threadIdx.x;
    if (idx >= Kseg * Ldim) return;
    int k = idx >> 6, l = idx & 63;
    float acc = cls_b[l];
    for (int d = 0; d < Ddim; ++d)
        acc = fmaf(cls_w[l * Ddim + d], conv_b[k * Ddim + d], acc);
    bias2[idx] = acc;
}

// ============================================================================
// Kernel 3: WMMA GEMM.  U = word_rep(f32->f16) @ W2h^T, f32 accum, f16 out.
// Block tile 128x128, BK=32, 8 waves of 64x32, v_wmma_f32_16x16x32_f16.
// Double-buffered LDS; both A and B tiles stored fragment-contiguous so every
// fragment is exactly 2x ds_load_b128.
// ============================================================================
#define BK   32
#define ASTR 40     // padded row stride in halves (80 B: keeps 16B alignment,
#define BSTR 40     // and 20*c mod 64 is distinct for c=0..15 -> no conflicts)
#define NKIT (Ddim / BK)   // 16 k-iterations

__global__ __launch_bounds__(256) void gemm_kernel(
    const float*    __restrict__ A,    // word_rep (Mrows x D)
    const _Float16* __restrict__ Bm,   // W2h      (Ncols x D)  (== B^T)
    _Float16*       __restrict__ U)    // (Mrows x Ncols)
{
    __shared__ _Float16 Ash[2][128 * ASTR];   // [buf][row][kk]
    __shared__ _Float16 Bsh[2][128 * BSTR];   // [buf][col][kk]  (straight copy!)

    int t    = threadIdx.x;
    int gn0  = blockIdx.x * 128;
    int gm0  = blockIdx.y * 128;
    int wid  = t >> 5, lane = t & 31;
    int hl   = lane & 15, hi = lane >> 4;
    int wm   = (wid >> 2) * 64;   // wave M origin: 0 / 64
    int wn   = (wid & 3) * 32;    // wave N origin: 0/32/64/96

    v8f acc[4][2];
#pragma unroll
    for (int mi = 0; mi < 4; ++mi)
#pragma unroll
        for (int ni = 0; ni < 2; ++ni)
#pragma unroll
            for (int r = 0; r < 8; ++r) acc[mi][ni][r] = 0.0f;

    // loader mapping: each thread moves 16 contiguous elements of one row
    int lrow = t >> 1;            // 0..127
    int lcg  = (t & 1) * 16;      // 0 / 16

    const float*    gA = A  + (size_t)(gm0 + lrow) * Ddim + lcg;
    const _Float16* gB = Bm + (size_t)(gn0 + lrow) * Ddim + lcg;

    // ---- stage tile 0 ----
    float4 a0 = ((const float4*)gA)[0];
    float4 a1 = ((const float4*)gA)[1];
    float4 a2 = ((const float4*)gA)[2];
    float4 a3 = ((const float4*)gA)[3];
    uint4  b0 = ((const uint4*)gB)[0];
    uint4  b1 = ((const uint4*)gB)[1];
    {
        __attribute__((aligned(16))) _Float16 ah[16];
        ah[0]=(_Float16)a0.x; ah[1]=(_Float16)a0.y; ah[2]=(_Float16)a0.z; ah[3]=(_Float16)a0.w;
        ah[4]=(_Float16)a1.x; ah[5]=(_Float16)a1.y; ah[6]=(_Float16)a1.z; ah[7]=(_Float16)a1.w;
        ah[8]=(_Float16)a2.x; ah[9]=(_Float16)a2.y; ah[10]=(_Float16)a2.z; ah[11]=(_Float16)a2.w;
        ah[12]=(_Float16)a3.x; ah[13]=(_Float16)a3.y; ah[14]=(_Float16)a3.z; ah[15]=(_Float16)a3.w;
        uint4* as = (uint4*)&Ash[0][lrow * ASTR + lcg];
        as[0] = ((const uint4*)ah)[0];
        as[1] = ((const uint4*)ah)[1];
        uint4* bs = (uint4*)&Bsh[0][lrow * BSTR + lcg];
        bs[0] = b0;
        bs[1] = b1;
    }
    __syncthreads();

    for (int it = 0; it < NKIT; ++it) {
        int cur = it & 1;
        bool more = (it + 1 < NKIT);
        // ---- issue next tile's global loads (overlap with WMMA below) ----
        if (more) {
            const float*    ga = gA + (it + 1) * BK;
            const _Float16* gb = gB + (it + 1) * BK;
            a0 = ((const float4*)ga)[0];
            a1 = ((const float4*)ga)[1];
            a2 = ((const float4*)ga)[2];
            a3 = ((const float4*)ga)[3];
            b0 = ((const uint4*)gb)[0];
            b1 = ((const uint4*)gb)[1];
        }
        // ---- WMMA on current buffer ----
        v16h bf[2];
#pragma unroll
        for (int ni = 0; ni < 2; ++ni) {
            int col = wn + ni * 16 + hl;
#pragma unroll
            for (int i = 0; i < 16; ++i)
                bf[ni][i] = Bsh[cur][col * BSTR + hi * 16 + i];   // contiguous: 2x b128
        }
#pragma unroll
        for (int mi = 0; mi < 4; ++mi) {
            v16h af;
            int row = wm + mi * 16 + hl;
#pragma unroll
            for (int i = 0; i < 8; ++i) {
                af[i]     = Ash[cur][row * ASTR + hi * 16 + i];
                af[i + 8] = Ash[cur][row * ASTR + hi * 16 + 8 + i];
            }
#pragma unroll
            for (int ni = 0; ni < 2; ++ni)
                acc[mi][ni] = __builtin_amdgcn_wmma_f32_16x16x32_f16(
                    false, af, false, bf[ni], (short)0, acc[mi][ni], false, false);
        }
        // ---- stage next tile into the other buffer ----
        if (more) {
            int nxt = cur ^ 1;
            __attribute__((aligned(16))) _Float16 ah[16];
            ah[0]=(_Float16)a0.x; ah[1]=(_Float16)a0.y; ah[2]=(_Float16)a0.z; ah[3]=(_Float16)a0.w;
            ah[4]=(_Float16)a1.x; ah[5]=(_Float16)a1.y; ah[6]=(_Float16)a1.z; ah[7]=(_Float16)a1.w;
            ah[8]=(_Float16)a2.x; ah[9]=(_Float16)a2.y; ah[10]=(_Float16)a2.z; ah[11]=(_Float16)a2.w;
            ah[12]=(_Float16)a3.x; ah[13]=(_Float16)a3.y; ah[14]=(_Float16)a3.z; ah[15]=(_Float16)a3.w;
            uint4* as = (uint4*)&Ash[nxt][lrow * ASTR + lcg];
            as[0] = ((const uint4*)ah)[0];
            as[1] = ((const uint4*)ah)[1];
            uint4* bs = (uint4*)&Bsh[nxt][lrow * BSTR + lcg];
            bs[0] = b0;
            bs[1] = b1;
        }
        __syncthreads();
    }

    // Epilogue: f32 C layout -> f16 U (VGPR r: M=r+8*hi, N=hl)
#pragma unroll
    for (int mi = 0; mi < 4; ++mi)
#pragma unroll
        for (int ni = 0; ni < 2; ++ni) {
            int n = gn0 + wn + ni * 16 + hl;
#pragma unroll
            for (int r = 0; r < 8; ++r) {
                int m = gm0 + wm + mi * 16 + hi * 8 + r;
                U[(size_t)m * Ncols + n] = (_Float16)acc[mi][ni][r];
            }
        }
}

// ============================================================================
// Kernel 4: label_score[b,s,k,l] = bias2[k,l] + sum_{j<=k} U[b, s-k+j, (k,j), l]
// ============================================================================
__global__ __launch_bounds__(256) void assemble_kernel(
    const _Float16* __restrict__ U,
    const float*    __restrict__ bias2,
    const int*      __restrict__ mask,
    float*          __restrict__ LS)    // (B,S,K,L)
{
    int bs = blockIdx.x;
    int b = bs >> 8, s = bs & 255;
    int mt = mask[b * Sdim + s];
    for (int e = threadIdx.x; e < Kseg * Ldim; e += blockDim.x) {
        int k = e >> 6, l = e & 63;
        float v;
        if (mt == 0 || k > s || l == 0) {
            v = NEGV;
        } else {
            v = bias2[e];
            int pk0 = k * (k + 1) / 2;
            for (int j = 0; j <= k; ++j) {
                size_t row = (size_t)b * Sdim + (s - k + j);
                v += (float)U[row * Ncols + (size_t)(pk0 + j) * Ldim + l];
            }
        }
        LS[((size_t)bs * Kseg + k) * Ldim + l] = v;
    }
}

// ============================================================================
// Kernel 5: semi-CRF forward scan.  One block per batch element.
// Rings in LDS:  R[age][l] = lse_k(row),  M[age][l] = lse_{l'}(T[l,l']+R[l']).
// ============================================================================
__global__ __launch_bounds__(256) void scan_kernel(
    const float* __restrict__ LS,
    const int*   __restrict__ mask,
    const float* __restrict__ T,
    const float* __restrict__ Tfb,
    const float* __restrict__ Tte,
    float*       __restrict__ out)
{
    __shared__ float T_sh[64][65];
    __shared__ float M_ring[Kseg][Ldim];
    __shared__ float R_ring[Kseg][Ldim];
    __shared__ float row_buf[Kseg][Ldim];
    __shared__ float G[Kseg][Ldim];
    __shared__ float tfb[Ldim], tte[Ldim];
    __shared__ float red[256];
    __shared__ int   ired[256];
    __shared__ int   s_len;
    __shared__ float s_mx;

    int b = blockIdx.x;
    int t = threadIdx.x;

    for (int e = t; e < 64 * 64; e += 256) T_sh[e >> 6][e & 63] = T[e];
    if (t < 64) { tfb[t] = Tfb[t]; tte[t] = Tte[t]; }
    for (int e = t; e < Kseg * Ldim; e += 256) {
        R_ring[e >> 6][e & 63] = NEGV;
        G[e >> 6][e & 63]      = NEGV;
    }
    ired[t] = mask[b * Sdim + t];
    __syncthreads();
    for (int off = 128; off > 0; off >>= 1) {
        if (t < off) ired[t] += ired[t + off];
        __syncthreads();
    }
    if (t == 0) s_len = ired[0];
    __syncthreads();
    int len = s_len;

    int lq = t >> 2, sub = t & 3;   // M step: 4 lanes per label row
    int kk = t >> 6, ll = t & 63;   // row step: (k, k+4) per thread

    for (int st = 0; st < Sdim; ++st) {
        // ---- M update: M[st-1] = lse_{l'}(T[l,l'] + R[st-1][l']) ----
        if (st >= 1) {
            int slot = (st - 1) & (Kseg - 1);
            float vbuf[16];
            float mx = -3.0e38f;
#pragma unroll
            for (int i = 0; i < 16; ++i) {
                float v = T_sh[lq][sub * 16 + i] + R_ring[slot][sub * 16 + i];
                vbuf[i] = v;
                mx = fmaxf(mx, v);
            }
            mx = fmaxf(mx, __shfl_xor(mx, 1));
            mx = fmaxf(mx, __shfl_xor(mx, 2));
            float sm = 0.0f;
#pragma unroll
            for (int i = 0; i < 16; ++i) sm += __expf(vbuf[i] - mx);
            sm += __shfl_xor(sm, 1);
            sm += __shfl_xor(sm, 2);
            if (sub == 0) M_ring[slot][lq] = mx + __logf(sm);
        }
        __syncthreads();

        // ---- rows: begin / rec / NEG per (k,l) ----
        int mt = mask[b * Sdim + st];
#pragma unroll
        for (int h = 0; h < 2; ++h) {
            int k = kk + h * 4;
            float ls = LS[(((size_t)b * Sdim + st) * Kseg + k) * Ldim + ll];
            float v;
            if (k > st)       v = NEGV;
            else if (k == st) v = ls + tfb[ll];
            else              v = mt ? (ls + M_ring[(st - 1 - k) & (Kseg - 1)][ll]) : NEGV;
            row_buf[k][ll] = v;
        }
        __syncthreads();

        // ---- R_new[l] = lse_k(row),  capture g at t == len-1-k ----
        if (t < 64) {
            float mx = row_buf[0][t];
#pragma unroll
            for (int k = 1; k < Kseg; ++k) mx = fmaxf(mx, row_buf[k][t]);
            float sm = 0.0f;
#pragma unroll
            for (int k = 0; k < Kseg; ++k) sm += __expf(row_buf[k][t] - mx);
            R_ring[st & (Kseg - 1)][t] = mx + __logf(sm);
            int cg = len - 1 - st;
            if (cg >= 0 && cg < Kseg) G[cg][t] = row_buf[cg][t];
        }
        __syncthreads();
    }

    // k with len-1-k < 0 falls back to rows[S-1] (still in row_buf)
    if (t < 64) {
        for (int k = 0; k < Kseg; ++k)
            if (len - 1 - k < 0) G[k][t] = row_buf[k][t];
    }
    __syncthreads();

    // ---- out[b] = logsumexp_{k,l}(G + T_to_END) ----
    float v0 = G[t >> 6][t & 63] + tte[t & 63];
    int t2 = t + 256;
    float v1 = G[t2 >> 6][t2 & 63] + tte[t2 & 63];
    red[t] = fmaxf(v0, v1);
    __syncthreads();
    for (int off = 128; off > 0; off >>= 1) {
        if (t < off) red[t] = fmaxf(red[t], red[t + off]);
        __syncthreads();
    }
    if (t == 0) s_mx = red[0];
    __syncthreads();
    float mx = s_mx;
    red[t] = __expf(v0 - mx) + __expf(v1 - mx);
    __syncthreads();
    for (int off = 128; off > 0; off >>= 1) {
        if (t < off) red[t] += red[t + off];
        __syncthreads();
    }
    if (t == 0) out[b] = mx + __logf(red[0]);
}

// ============================================================================
extern "C" void kernel_launch(void* const* d_in, const int* in_sizes, int n_in,
                              void* d_out, int out_size, void* d_ws, size_t ws_size,
                              hipStream_t stream)
{
    const float* word_rep = (const float*)d_in[0];   // (B,S,D)
    const int*   mask     = (const int*)  d_in[1];   // (B,S)
    const float* conv_w   = (const float*)d_in[2];   // (K,D,D,K)
    const float* conv_b   = (const float*)d_in[3];   // (K,D)
    const float* cls_w    = (const float*)d_in[4];   // (L,D)
    const float* cls_b    = (const float*)d_in[5];   // (L,)
    const float* T        = (const float*)d_in[6];   // (L,L)
    const float* Tfb      = (const float*)d_in[7];   // (L,)
    const float* Tte      = (const float*)d_in[8];   // (L,)
    float* out = (float*)d_out;                      // (B,)

    // Workspace layout (256B aligned slabs)
    char* ws = (char*)d_ws;
    _Float16* W2h   = (_Float16*)(ws);                                   //  2,359,296 B
    float*    bias2 = (float*)   (ws + 2359296);                         //      2,048 B
    _Float16* U     = (_Float16*)(ws + 2361344);                         // 37,748,736 B
    float*    LS    = (float*)   (ws + 40110080);                        // 16,777,216 B

    w2_kernel<<<dim3(NPAIR, Ddim / 64), 256, 0, stream>>>(conv_w, cls_w, W2h);
    bias2_kernel<<<2, 256, 0, stream>>>(cls_w, cls_b, conv_b, bias2);
    gemm_kernel<<<dim3(Ncols / 128, Mrows / 128), 256, 0, stream>>>(word_rep, W2h, U);
    assemble_kernel<<<Mrows, 256, 0, stream>>>(U, bias2, mask, LS);
    scan_kernel<<<Bdim, 256, 0, stream>>>(LS, mask, T, Tfb, Tte, out);
}
